// SparseAttention_46127948759425
// MI455X (gfx1250) — matrix operand order repaired and checked by reading, not compiled, not run on hardware
//
#include <hip/hip_runtime.h>
#include <hip/hip_bf16.h>

typedef __attribute__((ext_vector_type(16))) _Float16 v16h;
typedef __attribute__((ext_vector_type(8)))  _Float16 v8h;
typedef __attribute__((ext_vector_type(8)))  float    v8f;

union V16 { v16h v; v8h h[2]; };

__device__ __forceinline__ v8h  ld8 (const _Float16* p){ return *reinterpret_cast<const v8h *>(p); }
__device__ __forceinline__ v16h ld16(const _Float16* p){ return *reinterpret_cast<const v16h*>(p); }

#define WMMA_F16(a,b,c) __builtin_amdgcn_wmma_f32_16x16x32_f16(false,(a),false,(b),(short)0,(c),false,false)

// ---------------- conversion kernels ----------------
__global__ void cvt_f16_kernel(const float* __restrict__ in, _Float16* __restrict__ out, int n){
    int i = blockIdx.x * 256 + threadIdx.x;
    if (i < n) out[i] = (_Float16)in[i];
}

// in: rows x cols (row-major) -> out: cols x rows (row-major)  [i.e. W^T]
__global__ void cvtT_f16_kernel(const float* __restrict__ in, _Float16* __restrict__ out,
                                int rows, int cols){
    int i = blockIdx.x * 256 + threadIdx.x;
    if (i < rows * cols){
        int j = i / rows;            // output row = original column
        int r = i % rows;            // output col = original row
        out[(size_t)j * rows + r] = (_Float16)in[(size_t)r * cols + j];
    }
}

// ---------------- shared 32x64 strip GEMM core (K multiple of 32) ----------------
// A: MxK row-major f16; Wt: NxK row-major f16 (i.e. W transposed).
// acc[mi][t] is the 16x16 C tile at rows m0+mi*16, cols n0+t*16.
__device__ __forceinline__ void gemm_strip32(const _Float16* __restrict__ A,
                                             const _Float16* __restrict__ Wt,
                                             int K, int m0, int n0, int lane,
                                             v8f acc[2][4]){
    const int half = lane >> 4, ln = lane & 15;
    const _Float16* ap0 = A + (size_t)(m0 + ln) * K;
    const _Float16* ap1 = A + (size_t)(m0 + 16 + ln) * K;
    const _Float16* bp0 = Wt + (size_t)(n0 + ln) * K;
    for (int k = 0; k < K; k += 32){
        __builtin_prefetch(ap0 + k + 256, 0, 1);     // global_prefetch_b8
        __builtin_prefetch(bp0 + k + 256, 0, 1);
        V16 a0, a1;
        a0.h[0] = ld8(ap0 + k     + half * 16);
        a0.h[1] = ld8(ap0 + k + 8 + half * 16);
        a1.h[0] = ld8(ap1 + k     + half * 16);
        a1.h[1] = ld8(ap1 + k + 8 + half * 16);
#pragma unroll
        for (int t = 0; t < 4; ++t){
            v16h bf = ld16(Wt + (size_t)(n0 + t*16 + ln) * K + k + half * 16);
            acc[0][t] = WMMA_F16(a0.v, bf, acc[0][t]);
            acc[1][t] = WMMA_F16(a1.v, bf, acc[1][t]);
        }
    }
}

// ---------------- QKV GEMM:  [4096x1024] x [1024x3072] ----------------
// Wave computes a 32x64 strip.  Writes q[b,h,t,d], k[b,h,t,d], vT[b,h,d,t] (f16).
__global__ __launch_bounds__(128) void qkv_gemm_kernel(
        const _Float16* __restrict__ Xh, const _Float16* __restrict__ Wt,
        const float* __restrict__ bias,
        _Float16* __restrict__ qh, _Float16* __restrict__ kh, _Float16* __restrict__ vTh){
    const int lane = threadIdx.x & 31;
    const int wib  = threadIdx.x >> 5;
    const int gw   = blockIdx.x * 4 + wib;      // 6144 waves
    const int mt   = gw & 127;                  // 128 m-tiles of 32 rows
    const int ns   = gw >> 7;                   // 48 n-strips of 64
    const int m0 = mt * 32, n0 = ns * 64;
    const int half = lane >> 4, ln = lane & 15;

    v8f acc[2][4]; v8f z = {};
#pragma unroll
    for (int mi = 0; mi < 2; ++mi)
#pragma unroll
        for (int t = 0; t < 4; ++t) acc[mi][t] = z;

    gemm_strip32(Xh, Wt, 1024, m0, n0, lane, acc);

    const int s  = n0 / 1024;                   // 0=q 1=k 2=v
    const int hh = (n0 % 1024) / 64;            // head
#pragma unroll
    for (int t = 0; t < 4; ++t){
        const int col = n0 + t*16 + ln;
        const float bv = bias[col];
        const int dd = t*16 + ln;               // col % 64
#pragma unroll
        for (int mi = 0; mi < 2; ++mi)
#pragma unroll
            for (int v = 0; v < 8; ++v){
                const int r   = m0 + mi*16 + v + 8*half;  // global row in [0,4096)
                const int bb  = r >> 11;                  // batch
                const int tok = r & 2047;                 // token
                const _Float16 val = (_Float16)(acc[mi][t][v] + bv);
                const size_t bhd = (size_t)(bb * 16 + hh);
                if      (s == 0) qh [(bhd * 2048 + tok) * 64 + dd] = val;
                else if (s == 1) kh [(bhd * 2048 + tok) * 64 + dd] = val;
                else             vTh[(bhd * 64 + dd) * 2048 + tok] = val;
            }
    }
}

// ---------------- Flash attention: one wave per (b,h,32-query tile) ----------------
__global__ __launch_bounds__(128) void attn_kernel(
        const _Float16* __restrict__ qh, const _Float16* __restrict__ kh,
        const _Float16* __restrict__ vTh, _Float16* __restrict__ Ah){
    const int lane  = threadIdx.x & 31;
    const int wib   = threadIdx.x >> 5;
    const int gw    = blockIdx.x * 4 + wib;     // 2048 waves
    const int qtile = gw & 63;                  // 64 query tiles of 32
    const int bh    = gw >> 6;                  // b*16+h in [0,32)
    const int b = bh >> 4, h = bh & 15;
    const int q0 = qtile * 32;
    const int half = lane >> 4, ln = lane & 15;

    const _Float16* qbase = qh  + (size_t)bh * 2048 * 64;
    const _Float16* kbase = kh  + (size_t)bh * 2048 * 64;
    const _Float16* vbase = vTh + (size_t)bh * 64 * 2048;

    // Q A-fragments: [mtile][d-chunk], d-chunks [0,32) and [32,64)
    V16 aQ[2][2];
#pragma unroll
    for (int mi = 0; mi < 2; ++mi){
        const _Float16* qp = qbase + (size_t)(q0 + mi*16 + ln) * 64;
        const int o = half * 16;
        aQ[mi][0].h[0] = ld8(qp + o);        aQ[mi][0].h[1] = ld8(qp + 8 + o);
        aQ[mi][1].h[0] = ld8(qp + 32 + o);   aQ[mi][1].h[1] = ld8(qp + 40 + o);
    }

    float m[2][8], l[2][8];
    v8f accd[2][4]; v8f zero = {};
#pragma unroll
    for (int mi = 0; mi < 2; ++mi){
#pragma unroll
        for (int v = 0; v < 8; ++v){ m[mi][v] = -1e30f; l[mi][v] = 0.f; }
#pragma unroll
        for (int t = 0; t < 4; ++t) accd[mi][t] = zero;
    }

    __shared__ alignas(32) _Float16 plds[4][2][16 * 32];  // P tiles per wave
    const float scale = 0.5f;                             // 1/sqrt(d//h) = 1/sqrt(4)

    for (int kc = 0; kc < 2048; kc += 32){
        // ---- K B-fragments for 32 keys (two 16-key tiles x two d-chunks), shared by both m-tiles
        v16h bk[2][2];
#pragma unroll
        for (int kt = 0; kt < 2; ++kt){
            const _Float16* kp = kbase + (size_t)(kc + kt*16 + ln) * 64 + half * 16;
            bk[kt][0] = ld16(kp);        // d in [0,32)
            bk[kt][1] = ld16(kp + 32);   // d in [32,64)
        }
#pragma unroll
        for (int mi = 0; mi < 2; ++mi){
            // ---- S = Q K^T
            v8f s0 = WMMA_F16(aQ[mi][0].v, bk[0][0], zero);
            s0     = WMMA_F16(aQ[mi][1].v, bk[0][1], s0);
            v8f s1 = WMMA_F16(aQ[mi][0].v, bk[1][0], zero);
            s1     = WMMA_F16(aQ[mi][1].v, bk[1][1], s1);
            // ---- online softmax (row = v + 8*half, cols across lanes)
            float p0[8], p1[8], corr[8];
#pragma unroll
            for (int v = 0; v < 8; ++v){
                float x0 = s0[v] * scale, x1 = s1[v] * scale;
                float mx = fmaxf(x0, x1);
#pragma unroll
                for (int off = 1; off < 16; off <<= 1) mx = fmaxf(mx, __shfl_xor(mx, off, 32));
                const float mn = fmaxf(m[mi][v], mx);
                const float c  = __expf(m[mi][v] - mn);
                const float e0 = __expf(x0 - mn), e1 = __expf(x1 - mn);
                float rs = e0 + e1;
#pragma unroll
                for (int off = 1; off < 16; off <<= 1) rs += __shfl_xor(rs, off, 32);
                l[mi][v] = l[mi][v] * c + rs;
                m[mi][v] = mn;
                p0[v] = e0; p1[v] = e1; corr[v] = c;
            }
#pragma unroll
            for (int t = 0; t < 4; ++t)
#pragma unroll
                for (int v = 0; v < 8; ++v) accd[mi][t][v] *= corr[v];
            // ---- stage P (16x32) through LDS: C-layout -> A-fragment layout
#pragma unroll
            for (int v = 0; v < 8; ++v){
                const int row = v + 8 * half;
                plds[wib][mi][row * 32 + ln]      = (_Float16)p0[v];
                plds[wib][mi][row * 32 + 16 + ln] = (_Float16)p1[v];
            }
        }
        asm volatile("s_wait_dscnt 0" ::: "memory");
        V16 aP[2];
#pragma unroll
        for (int mi = 0; mi < 2; ++mi){
            aP[mi].h[0] = *reinterpret_cast<const v8h*>(&plds[wib][mi][ln * 32 + half * 16]);
            aP[mi].h[1] = *reinterpret_cast<const v8h*>(&plds[wib][mi][ln * 32 + half * 16 + 8]);
        }
        // ---- acc += P @ V  (V read transposed: contiguous per-lane loads, shared by m-tiles)
#pragma unroll
        for (int t = 0; t < 4; ++t){
            const _Float16* vp = vbase + (size_t)(t*16 + ln) * 2048 + kc + half * 16;
            v16h bv = ld16(vp);
            accd[0][t] = WMMA_F16(aP[0].v, bv, accd[0][t]);
            accd[1][t] = WMMA_F16(aP[1].v, bv, accd[1][t]);
        }
    }

    // ---- epilogue: normalize and scatter to [b, t, h*64+d] (f16)
#pragma unroll
    for (int mi = 0; mi < 2; ++mi)
#pragma unroll
        for (int t = 0; t < 4; ++t)
#pragma unroll
            for (int v = 0; v < 8; ++v){
                const int row = q0 + mi*16 + v + 8*half;
                const int col = h*64 + t*16 + ln;
                const float val = accd[mi][t][v] / l[mi][v];
                Ah[((size_t)(b * 2048 + row)) * 1024 + col] = (_Float16)val;
            }
}

// ---------------- Proj GEMM: [4096x1024] x [1024x1024] + bias -> fp32 out ----------------
__global__ __launch_bounds__(128) void proj_gemm_kernel(
        const _Float16* __restrict__ Ah, const _Float16* __restrict__ Wt,
        const float* __restrict__ bias, float* __restrict__ out){
    const int lane = threadIdx.x & 31;
    const int wib  = threadIdx.x >> 5;
    const int gw   = blockIdx.x * 4 + wib;      // 2048 waves
    const int mt   = gw & 127;                  // 128 m-tiles of 32 rows
    const int ns   = gw >> 7;                   // 16 n-strips of 64
    const int m0 = mt * 32, n0 = ns * 64;
    const int half = lane >> 4, ln = lane & 15;

    v8f acc[2][4]; v8f z = {};
#pragma unroll
    for (int mi = 0; mi < 2; ++mi)
#pragma unroll
        for (int t = 0; t < 4; ++t) acc[mi][t] = z;

    gemm_strip32(Ah, Wt, 1024, m0, n0, lane, acc);

#pragma unroll
    for (int t = 0; t < 4; ++t){
        const int col = n0 + t*16 + ln;
        const float bv = bias[col];
#pragma unroll
        for (int mi = 0; mi < 2; ++mi)
#pragma unroll
            for (int v = 0; v < 8; ++v){
                const int r = m0 + mi*16 + v + 8*half;
                out[(size_t)r * 1024 + col] = acc[mi][t][v] + bv;
            }
    }
}

// ---------------- host launch ----------------
extern "C" void kernel_launch(void* const* d_in, const int* in_sizes, int n_in,
                              void* d_out, int out_size, void* d_ws, size_t ws_size,
                              hipStream_t stream) {
    (void)in_sizes; (void)n_in; (void)out_size; (void)ws_size;
    const float* x     = (const float*)d_in[0];   // (2,2048,1024)
    const float* Wqkv  = (const float*)d_in[1];   // (1024,3072)
    const float* bqkv  = (const float*)d_in[2];   // (3072,)
    const float* Wproj = (const float*)d_in[3];   // (1024,1024)
    const float* bproj = (const float*)d_in[4];   // (1024,)
    float* out = (float*)d_out;

    char* ws = (char*)d_ws;
    size_t off = 0;
    auto take = [&](size_t bytes) -> void* {
        void* p = ws + off;
        off += (bytes + 255) & ~(size_t)255;
        return p;
    };
    _Float16* Xh     = (_Float16*)take((size_t)4096 * 1024 * 2);
    _Float16* WqkvT  = (_Float16*)take((size_t)3072 * 1024 * 2);
    _Float16* WprojT = (_Float16*)take((size_t)1024 * 1024 * 2);
    _Float16* qh     = (_Float16*)take((size_t)4096 * 1024 * 2);
    _Float16* kh     = (_Float16*)take((size_t)4096 * 1024 * 2);
    _Float16* vTh    = (_Float16*)take((size_t)4096 * 1024 * 2);
    _Float16* Ah     = (_Float16*)take((size_t)4096 * 1024 * 2);

    cvt_f16_kernel <<<(4096*1024)/256, 256, 0, stream>>>(x, Xh, 4096*1024);
    cvtT_f16_kernel<<<(3072*1024)/256, 256, 0, stream>>>(Wqkv,  WqkvT,  1024, 3072);
    cvtT_f16_kernel<<<(1024*1024)/256, 256, 0, stream>>>(Wproj, WprojT, 1024, 1024);

    qkv_gemm_kernel <<<1536, 128, 0, stream>>>(Xh, WqkvT, bqkv, qh, kh, vTh);
    attn_kernel     <<< 512, 128, 0, stream>>>(qh, kh, vTh, Ah);
    proj_gemm_kernel<<< 512, 128, 0, stream>>>(Ah, WprojT, bproj, out);
}
